// PointNetPPEncoder_31507880083701
// MI455X (gfx1250) — matrix operand order repaired and checked by reading, not compiled, not run on hardware
//
#include <hip/hip_runtime.h>
#include <hip/hip_bf16.h>
#include <cstdint>

typedef __attribute__((ext_vector_type(16))) _Float16     v16h;
typedef __attribute__((ext_vector_type(8)))  float        v8f;
typedef __attribute__((ext_vector_type(8)))  unsigned int v8u;
typedef __attribute__((ext_vector_type(4)))  unsigned int v4u;

#define THREADS 256
#define ROWS 32   // GEMM rows per workgroup (2 WMMA row-tiles)

// ---------------------------------------------------------------------------
// Farthest point sampling: one workgroup per batch, sequential scan with
// block-wide argmax (tie-break: lowest index, matching jnp.argmax).
// ---------------------------------------------------------------------------
__global__ void fps_kernel(const float* __restrict__ xyz, int N, int np,
                           int* __restrict__ cidx)
{
  __shared__ float dist[4096];
  __shared__ float rb[THREADS];
  __shared__ int   rbi[THREADS];
  __shared__ int   far_s;
  const int b   = blockIdx.x;
  const int tid = threadIdx.x;
  const float* px = xyz + (size_t)b * N * 3;

  for (int i = tid; i < N; i += THREADS) dist[i] = 1e10f;
  if (tid == 0) far_s = 0;
  __syncthreads();

  for (int t = 0; t < np; ++t) {
    const int f = far_s;
    if (tid == 0) cidx[b * np + t] = f;
    const float cx = px[f*3+0], cy = px[f*3+1], cz = px[f*3+2];
    float lbest = -1.0f; int lbi = 0x7fffffff;
    for (int i = tid; i < N; i += THREADS) {
      float dx = px[i*3+0]-cx, dy = px[i*3+1]-cy, dz = px[i*3+2]-cz;
      float d  = dx*dx + dy*dy + dz*dz;
      float nd = fminf(dist[i], d);
      dist[i] = nd;
      if (nd > lbest) { lbest = nd; lbi = i; }
    }
    rb[tid] = lbest; rbi[tid] = lbi;
    __syncthreads();
    for (int s = THREADS/2; s > 0; s >>= 1) {
      if (tid < s) {
        float a = rb[tid], c = rb[tid+s];
        if (c > a || (c == a && rbi[tid+s] < rbi[tid])) { rb[tid] = c; rbi[tid] = rbi[tid+s]; }
      }
      __syncthreads();
    }
    if (tid == 0) far_s = rbi[0];
    __syncthreads();
  }
}

// ---------------------------------------------------------------------------
// kNN (iterative masked argmin == stable top_k of -dist) + gather/center/concat,
// emitting f16 GEMM rows padded to cpad channels. One workgroup per centroid.
// ---------------------------------------------------------------------------
__global__ void knn_group_kernel(const float* __restrict__ xyz,
                                 const float* __restrict__ feats,
                                 const int*  __restrict__ cidx,
                                 float*      __restrict__ cent_out,
                                 _Float16*   __restrict__ X,
                                 int N, int np, int k, int Cf, int cpad)
{
  __shared__ float dist[4096];
  __shared__ float rb[THREADS];
  __shared__ int   rbi[THREADS];
  __shared__ int   nnidx[64];
  const int g   = blockIdx.x;          // b*np + n
  const int b   = g / np;
  const int tid = threadIdx.x;
  const float* px = xyz + (size_t)b * N * 3;
  const int ci = cidx[g];
  const float cx = px[ci*3+0], cy = px[ci*3+1], cz = px[ci*3+2];
  if (tid == 0) { cent_out[g*3+0] = cx; cent_out[g*3+1] = cy; cent_out[g*3+2] = cz; }

  for (int i = tid; i < N; i += THREADS) {
    float dx = px[i*3+0]-cx, dy = px[i*3+1]-cy, dz = px[i*3+2]-cz;
    dist[i] = dx*dx + dy*dy + dz*dz;
  }
  __syncthreads();

  for (int kk = 0; kk < k; ++kk) {
    float lbest = 3.0e38f; int lbi = 0x7fffffff;
    for (int i = tid; i < N; i += THREADS) {
      float d = dist[i];
      if (d < lbest) { lbest = d; lbi = i; }
    }
    rb[tid] = lbest; rbi[tid] = lbi;
    __syncthreads();
    for (int s = THREADS/2; s > 0; s >>= 1) {
      if (tid < s) {
        float a = rb[tid], c = rb[tid+s];
        if (c < a || (c == a && rbi[tid+s] < rbi[tid])) { rb[tid] = c; rbi[tid] = rbi[tid+s]; }
      }
      __syncthreads();
    }
    if (tid == 0) { nnidx[kk] = rbi[0]; dist[rbi[0]] = 3.4e38f; }
    __syncthreads();
  }

  const int total = k * cpad;
  _Float16* Xrow = X + (size_t)g * k * cpad;
  for (int e = tid; e < total; e += THREADS) {
    int kk = e / cpad, c = e % cpad;
    int src = nnidx[kk];
    float val = 0.f;
    if (c < 3)            val = px[src*3+c] - ((c==0)?cx:((c==1)?cy:cz));
    else if (c < 3 + Cf)  val = feats[((size_t)b*N + src)*Cf + (c-3)];
    Xrow[e] = (_Float16)val;
  }
}

// ---------------------------------------------------------------------------
// Weight prep: W (cout,cin) f32 -> Wt (cout, cpad) f16 K-major (zero-padded),
// fold BatchNorm into per-channel scale s and shift t.
// ---------------------------------------------------------------------------
__global__ void prep_kernel(const float* __restrict__ W,  const float* __restrict__ bb,
                            const float* __restrict__ g,  const float* __restrict__ be,
                            const float* __restrict__ m,  const float* __restrict__ v,
                            int cin, int cpad, int cout,
                            _Float16* __restrict__ Wt, float* __restrict__ s, float* __restrict__ t)
{
  int i = blockIdx.x * blockDim.x + threadIdx.x;
  if (i < cout) {
    float sc = g[i] * rsqrtf(v[i] + 1e-5f);
    s[i] = sc;
    t[i] = (bb[i] - m[i]) * sc + be[i];
  }
  int total = cpad * cout;
  if (i < total) {
    int o = i / cpad, kk = i % cpad;
    float w = (kk < cin) ? W[(size_t)o * cin + kk] : 0.f;
    Wt[i] = (_Float16)w;
  }
}

// ---------------------------------------------------------------------------
// WMMA fragment loaders (contiguous k = hs*16 + s order, same for A and B).
// A (16x32 f16): lane m=L%16; K = hs*16 + [0..15]
// -> one contiguous 32B LDS chunk per lane.
// B (32x16 f16, stored K-major per column n): lane n=L%16; K = hs*16 + [0..15]
// -> one contiguous 32B global chunk per lane.
// ---------------------------------------------------------------------------
__device__ __forceinline__ v16h load_fragA(const _Float16* rowBase, int stride,
                                           int k0, int lane)
{
  int mm = lane & 15, hs = lane >> 4;
  const _Float16* p = rowBase + (size_t)mm * stride + k0;
  v8u c = *(const v8u*)(p + hs * 16);
  return __builtin_bit_cast(v16h, c);
}

__device__ __forceinline__ v16h load_fragB(const _Float16* W, int kstride,
                                           int k0, int n0, int lane)
{
  int n = lane & 15, hs = lane >> 4;
  v8u c = *(const v8u*)(W + (size_t)(n0 + n) * kstride + k0 + hs * 16);
  return __builtin_bit_cast(v16h, c);
}

// GEMM (ROWS x Kdim) * (Kdim x cout) + BN + ReLU -> f16 LDS buffer (stride cout)
__device__ __forceinline__ void gemm_relu_store(const _Float16* inBuf, int Kdim,
                                                const _Float16* W,
                                                const float* __restrict__ s,
                                                const float* __restrict__ t,
                                                int cout, _Float16* outBuf,
                                                int wave, int lane)
{
  int ntiles = cout >> 4;
  for (int tile = wave; tile < 2 * ntiles; tile += THREADS / 32) {
    int mt = tile & 1, nt = tile >> 1;
    v8f acc = {};
    for (int k0 = 0; k0 < Kdim; k0 += 32) {
      v16h a = load_fragA(inBuf + mt * 16 * Kdim, Kdim, k0, lane);
      v16h bfr = load_fragB(W, Kdim, k0, nt * 16, lane);
      acc = __builtin_amdgcn_wmma_f32_16x16x32_f16(false, a, false, bfr,
                                                   (short)0, acc, false, false);
    }
    int n = lane & 15, hs = lane >> 4;
    float sc = s[nt*16 + n], sh = t[nt*16 + n];
#pragma unroll
    for (int i = 0; i < 8; ++i) {
      float y = fmaf(acc[i], sc, sh);
      y = y > 0.f ? y : 0.f;
      outBuf[(size_t)(mt*16 + i + hs*8) * cout + nt*16 + n] = (_Float16)y;
    }
  }
}

// Final GEMM stage + BN + ReLU + max over the k neighbor rows -> global atomic
__device__ __forceinline__ void gemm_relu_max(const _Float16* inBuf, int Kdim,
                                              const _Float16* W,
                                              const float* __restrict__ s,
                                              const float* __restrict__ t,
                                              int cout, int rowBase, int k,
                                              float* __restrict__ feat,
                                              int wave, int lane)
{
  int ntiles = cout >> 4;
  for (int tile = wave; tile < 2 * ntiles; tile += THREADS / 32) {
    int mt = tile & 1, nt = tile >> 1;
    v8f acc = {};
    for (int k0 = 0; k0 < Kdim; k0 += 32) {
      v16h a = load_fragA(inBuf + mt * 16 * Kdim, Kdim, k0, lane);
      v16h bfr = load_fragB(W, Kdim, k0, nt * 16, lane);
      acc = __builtin_amdgcn_wmma_f32_16x16x32_f16(false, a, false, bfr,
                                                   (short)0, acc, false, false);
    }
    int n = lane & 15, hs = lane >> 4;
    float sc = s[nt*16 + n], sh = t[nt*16 + n];
    float cm = 0.f;
#pragma unroll
    for (int i = 0; i < 8; ++i) {
      float y = fmaf(acc[i], sc, sh);
      y = y > 0.f ? y : 0.f;
      cm = fmaxf(cm, y);
    }
    cm = fmaxf(cm, __shfl_xor(cm, 16, 32));      // fold the two half-waves
    if (hs == 0) {
      int centroid = (rowBase + mt * 16) / k;    // k in {16,32,64}: tiles aligned
      atomicMax((unsigned int*)&feat[(size_t)centroid * cout + nt*16 + n],
                __float_as_uint(cm));            // valid: ReLU => values >= 0
    }
  }
}

// ---------------------------------------------------------------------------
// Fused 3-stage MLP over 32 GEMM rows: intermediates stay in LDS.
// ---------------------------------------------------------------------------
__global__ void mlp_kernel(const _Float16* __restrict__ X, int cpad,
                           const _Float16* __restrict__ W1, const float* s1, const float* t1, int c1,
                           const _Float16* __restrict__ W2, const float* s2, const float* t2, int c2,
                           const _Float16* __restrict__ W3, const float* s3, const float* t3, int c3,
                           int k, float* __restrict__ feat)
{
  __shared__ __align__(32) _Float16 sIn[ROWS * 288];
  __shared__ __align__(32) _Float16 sA [ROWS * 256];
  __shared__ __align__(32) _Float16 sB [ROWS * 256];
  const int tid  = threadIdx.x;
  const int wave = tid >> 5, lane = tid & 31;
  const int rowBase = blockIdx.x * ROWS;

  { // cooperative 128-bit staging of input rows into LDS
    const v4u* src = (const v4u*)(X + (size_t)rowBase * cpad);
    v4u* dst = (v4u*)sIn;
    const int n8 = ROWS * cpad / 8;
    for (int i = tid; i < n8; i += THREADS) dst[i] = src[i];
  }
  __syncthreads();
  gemm_relu_store(sIn, cpad, W1, s1, t1, c1, sA, wave, lane);
  __syncthreads();
  gemm_relu_store(sA, c1, W2, s2, t2, c2, sB, wave, lane);
  __syncthreads();
  gemm_relu_max(sB, c2, W3, s3, t3, c3, rowBase, k, feat, wave, lane);
}

// Final max over npoints: out[b][c] = max_n feat[b][n][c]
__global__ void final_max_kernel(const float* __restrict__ feat,
                                 float* __restrict__ out, int np, int C)
{
  int i = blockIdx.x * blockDim.x + threadIdx.x;   // over B*C
  int b = i / C, c = i % C;
  float mx = 0.f;
  for (int n = 0; n < np; ++n) mx = fmaxf(mx, feat[((size_t)b * np + n) * C + c]);
  out[i] = mx;
}

// ---------------------------------------------------------------------------
extern "C" void kernel_launch(void* const* d_in, const int* in_sizes, int n_in,
                              void* d_out, int out_size, void* d_ws, size_t ws_size,
                              hipStream_t stream)
{
  (void)in_sizes; (void)n_in; (void)out_size; (void)ws_size;
  const float* xyz = (const float*)d_in[0];
  const int B = 8;

  uintptr_t base = (uintptr_t)d_ws;
  size_t off = 0;
  auto alloc = [&](size_t bytes) -> void* {
    off = (off + 255) & ~(size_t)255;
    void* p = (void*)(base + off);
    off += bytes;
    return p;
  };

  // Layer buffers
  int*       cidx1 = (int*)      alloc((size_t)B*1024*4);
  float*     cent1 = (float*)    alloc((size_t)B*1024*3*4);
  _Float16*  X1    = (_Float16*) alloc((size_t)B*1024*16*32*2);
  float*     feat1 = (float*)    alloc((size_t)B*1024*128*4);
  int*       cidx2 = (int*)      alloc((size_t)B*256*4);
  float*     cent2 = (float*)    alloc((size_t)B*256*3*4);
  _Float16*  X2    = (_Float16*) alloc((size_t)B*256*32*160*2);
  float*     feat2 = (float*)    alloc((size_t)B*256*256*4);
  int*       cidx3 = (int*)      alloc((size_t)B*64*4);
  float*     cent3 = (float*)    alloc((size_t)B*64*3*4);
  _Float16*  X3    = (_Float16*) alloc((size_t)B*64*64*288*2);
  float*     feat3 = (float*)    alloc((size_t)B*64*512*4);

  // Weight prep: 9 sublayers, (W,b,g,be,m,v) leaves flattened after xyz
  const int cins [9] = {3,64,64,  131,128,128,  259,256,256};
  const int cpads[9] = {32,64,64, 160,128,128,  288,256,256};
  const int couts[9] = {64,64,128,128,128,256,  256,256,512};
  _Float16* Wt[9]; float* ss[9]; float* tt[9];
  for (int i = 0; i < 9; ++i) {
    Wt[i] = (_Float16*)alloc((size_t)couts[i]*cpads[i]*2);
    ss[i] = (float*)   alloc((size_t)couts[i]*4);
    tt[i] = (float*)   alloc((size_t)couts[i]*4);
    int bi = 1 + i*6;
    int total = couts[i]*cpads[i];
    prep_kernel<<<(total + 255)/256, 256, 0, stream>>>(
        (const float*)d_in[bi+0], (const float*)d_in[bi+1], (const float*)d_in[bi+2],
        (const float*)d_in[bi+3], (const float*)d_in[bi+4], (const float*)d_in[bi+5],
        cins[i], cpads[i], couts[i], Wt[i], ss[i], tt[i]);
  }

  hipMemsetAsync(feat1, 0, (size_t)B*1024*128*4, stream);
  hipMemsetAsync(feat2, 0, (size_t)B*256*256*4, stream);
  hipMemsetAsync(feat3, 0, (size_t)B*64*512*4, stream);

  // SA layer 1: N=4096 -> np=1024, k=16, cin 3 (pad 32), mlp 64,64,128
  fps_kernel<<<B, THREADS, 0, stream>>>(xyz, 4096, 1024, cidx1);
  knn_group_kernel<<<B*1024, THREADS, 0, stream>>>(xyz, nullptr, cidx1, cent1, X1,
                                                   4096, 1024, 16, 0, 32);
  mlp_kernel<<<(B*1024*16)/ROWS, THREADS, 0, stream>>>(X1, 32,
      Wt[0],ss[0],tt[0],64,  Wt[1],ss[1],tt[1],64,  Wt[2],ss[2],tt[2],128, 16, feat1);

  // SA layer 2: N=1024 -> np=256, k=32, cin 131 (pad 160), mlp 128,128,256
  fps_kernel<<<B, THREADS, 0, stream>>>(cent1, 1024, 256, cidx2);
  knn_group_kernel<<<B*256, THREADS, 0, stream>>>(cent1, feat1, cidx2, cent2, X2,
                                                  1024, 256, 32, 128, 160);
  mlp_kernel<<<(B*256*32)/ROWS, THREADS, 0, stream>>>(X2, 160,
      Wt[3],ss[3],tt[3],128, Wt[4],ss[4],tt[4],128, Wt[5],ss[5],tt[5],256, 32, feat2);

  // SA layer 3: N=256 -> np=64, k=64, cin 259 (pad 288), mlp 256,256,512
  fps_kernel<<<B, THREADS, 0, stream>>>(cent2, 256, 64, cidx3);
  knn_group_kernel<<<B*64, THREADS, 0, stream>>>(cent2, feat2, cidx3, cent3, X3,
                                                 256, 64, 64, 256, 288);
  mlp_kernel<<<(B*64*64)/ROWS, THREADS, 0, stream>>>(X3, 288,
      Wt[6],ss[6],tt[6],256, Wt[7],ss[7],tt[7],256, Wt[8],ss[8],tt[8],512, 64, feat3);

  final_max_kernel<<<(B*512 + 255)/256, 256, 0, stream>>>(feat3, (float*)d_out, 64, 512);
}
